// FastRCNNHead_39642548142737
// MI455X (gfx1250) — matrix operand order repaired and checked by reading, not compiled, not run on hardware
//
#include <hip/hip_runtime.h>
#include <hip/hip_bf16.h>

typedef __attribute__((ext_vector_type(16))) _Float16 v16h;
typedef __attribute__((ext_vector_type(8)))  _Float16 v8h;
typedef __attribute__((ext_vector_type(8)))  float    v8f;

#define RES   7
#define NBOX  256
#define BATCH 2
#define CCH   256
#define FCDIM 1024
#define K1    (CCH * RES * RES)   /* 12544 */
#define OUTCH 93
#define KSPLIT 4

// -------------------------------------------------------------------------
// Kernel 1: ROI-align (level select by anchor//3), 14x14 samples -> 7x7 avg,
// store pooled activations as f16 in [M, K1] row-major (k = c*49 + ry*7 + rx).
// -------------------------------------------------------------------------
__global__ __launch_bounds__(256)
void roi_pool_kernel(const float* __restrict__ f0, const float* __restrict__ f1,
                     const float* __restrict__ f2, const float* __restrict__ f3,
                     const float* __restrict__ bbox, const int* __restrict__ anchor,
                     _Float16* __restrict__ pooled)
{
    __shared__ int   sy0[14], sy1[14], sx0[14], sx1[14];
    __shared__ float swy[14], swx[14];
    __shared__ const float* sfeat;
    __shared__ int   sH;

    const int m   = blockIdx.x;          // b*NBOX + n
    const int bat = m / NBOX;
    const int tid = threadIdx.x;

    if (tid < 28) {
        float y = bbox[m * 4 + 0], x = bbox[m * 4 + 1];
        float h = bbox[m * 4 + 2], w = bbox[m * 4 + 3];
        int   L = anchor[m] / 3;
        int   H = 128 >> L;
        float scale = 1.0f / (float)(4 << L);
        float tt = (y - 0.5f * h) * scale - 0.5f;
        float ll = (x - 0.5f * w) * scale - 0.5f;
        float btm = (y + 0.5f * h) * scale - 0.5f;
        float rr  = (x + 0.5f * w) * scale - 0.5f;
        if (tid == 0) {
            const float* fp[4] = { f0, f1, f2, f3 };
            sfeat = fp[L] + (size_t)bat * CCH * H * H;
            sH = H;
        }
        if (tid < 14) {
            float binh = (btm - tt) * (1.0f / RES);
            float yy = tt + binh * ((float)tid + 0.5f) * 0.5f;
            yy = fminf(fmaxf(yy, 0.0f), (float)(H - 1));
            int y0 = (int)floorf(yy);
            sy0[tid] = y0;
            sy1[tid] = (y0 + 1 < H - 1) ? (y0 + 1) : (H - 1);
            swy[tid] = yy - (float)y0;
        } else {
            int   q = tid - 14;
            float binw = (rr - ll) * (1.0f / RES);
            float xx = ll + binw * ((float)q + 0.5f) * 0.5f;
            xx = fminf(fmaxf(xx, 0.0f), (float)(H - 1));
            int x0 = (int)floorf(xx);
            sx0[q] = x0;
            sx1[q] = (x0 + 1 < H - 1) ? (x0 + 1) : (H - 1);
            swx[q] = xx - (float)x0;
        }
    }
    __syncthreads();

    const int c = tid;                       // one channel per thread (C == 256)
    const int H = sH;
    const float* fp = sfeat + (size_t)c * H * H;
    _Float16* orow = pooled + (size_t)m * K1 + (size_t)c * (RES * RES);

    for (int ry = 0; ry < RES; ++ry) {
        for (int rx = 0; rx < RES; ++rx) {
            float acc = 0.0f;
#pragma unroll
            for (int dy = 0; dy < 2; ++dy) {
                int   py = 2 * ry + dy;
                int   y0 = sy0[py], y1 = sy1[py];
                float fy = swy[py];
                const float* r0 = fp + (size_t)y0 * H;
                const float* r1 = fp + (size_t)y1 * H;
#pragma unroll
                for (int dx = 0; dx < 2; ++dx) {
                    int   px = 2 * rx + dx;
                    int   x0 = sx0[px], x1 = sx1[px];
                    float fx = swx[px];
                    float v = r0[x0] * (1.0f - fy) * (1.0f - fx)
                            + r0[x1] * (1.0f - fy) * fx
                            + r1[x0] * fy * (1.0f - fx)
                            + r1[x1] * fy * fx;
                    acc += v;
                }
            }
            orow[ry * RES + rx] = (_Float16)(acc * 0.25f);
        }
    }
}

// -------------------------------------------------------------------------
// Kernel 2: f32 -> f16 conversion (weights), grid-stride.
// -------------------------------------------------------------------------
__global__ void cvt_f32_to_f16(const float* __restrict__ x, _Float16* __restrict__ y, int n)
{
    int i = blockIdx.x * blockDim.x + threadIdx.x;
    int stride = gridDim.x * blockDim.x;
    for (; i < n; i += stride) y[i] = (_Float16)x[i];
}

// -------------------------------------------------------------------------
// Kernel 3: f16 WMMA GEMM with split-K; K and KCHUNK are compile-time so the
// K-loop has a static trip count (unroll + immediate-offset clauses, no v_nop
// filler).  Cpart[z][M,N] f32 = A[M, kz..kz+KCHUNK] @ Bw[N, kz..kz+KCHUNK]^T,
// z = blockIdx.z. One wave per 32x32 tile; 4x v_wmma_f32_16x16x32_f16 per
// K-step of 32.
// Fragment layouts per CDNA5 ISA 7.12.2:
//   A lanes 0-15 hold K={0..7,16..23}, lanes 16-31 hold K={8..15,24..31}
//   B lanes 0-15 hold K={0..7,16..23}, lanes 16-31 hold K={8..15,24..31}
//   D vgpr r: row = halfsel*8 + r, col = lane%16
// -------------------------------------------------------------------------
template <int K, int KCHUNK, int N>
__global__ __launch_bounds__(32)
void gemm_f16_wmma(const _Float16* __restrict__ A,   // [M,K]
                   const _Float16* __restrict__ Bw,  // [N,K]
                   float* __restrict__ Cpart,        // [KSPLIT][M,N]
                   int M)
{
    const int lane = threadIdx.x;
    const int hh   = lane >> 4;   // half-wave select
    const int l    = lane & 15;
    const int n0   = blockIdx.x * 32;
    const int m0   = blockIdx.y * 32;
    const int kbeg = blockIdx.z * KCHUNK;
    float* C = Cpart + (size_t)blockIdx.z * M * N;

    v8f acc[2][2] = {};
    union V { v16h v; v8h h[2]; };

    const _Float16* a0 = A  + (size_t)(m0 + l)      * K + kbeg + hh * 8;
    const _Float16* a1 = A  + (size_t)(m0 + 16 + l) * K + kbeg + hh * 8;
    const _Float16* b0 = Bw + (size_t)(n0 + l)      * K + kbeg + hh * 8;
    const _Float16* b1 = Bw + (size_t)(n0 + 16 + l) * K + kbeg + hh * 8;

#pragma unroll 4
    for (int kk = 0; kk < KCHUNK; kk += 32) {
        if (kk + 128 < KCHUNK) {            // gfx1250 global_prefetch_b8
            __builtin_prefetch(a0 + kk + 128, 0, 3);
            __builtin_prefetch(b0 + kk + 128, 0, 3);
        }
        V av0, av1, bv0, bv1;
        av0.h[0] = *(const v8h*)(a0 + kk);
        av0.h[1] = *(const v8h*)(a0 + kk + 16);
        av1.h[0] = *(const v8h*)(a1 + kk);
        av1.h[1] = *(const v8h*)(a1 + kk + 16);
        bv0.h[0] = *(const v8h*)(b0 + kk);
        bv0.h[1] = *(const v8h*)(b0 + kk + 16);
        bv1.h[0] = *(const v8h*)(b1 + kk);
        bv1.h[1] = *(const v8h*)(b1 + kk + 16);

        acc[0][0] = __builtin_amdgcn_wmma_f32_16x16x32_f16(false, av0.v, false, bv0.v,
                                                           (short)0, acc[0][0], false, false);
        acc[0][1] = __builtin_amdgcn_wmma_f32_16x16x32_f16(false, av0.v, false, bv1.v,
                                                           (short)0, acc[0][1], false, false);
        acc[1][0] = __builtin_amdgcn_wmma_f32_16x16x32_f16(false, av1.v, false, bv0.v,
                                                           (short)0, acc[1][0], false, false);
        acc[1][1] = __builtin_amdgcn_wmma_f32_16x16x32_f16(false, av1.v, false, bv1.v,
                                                           (short)0, acc[1][1], false, false);
    }

#pragma unroll
    for (int mi = 0; mi < 2; ++mi)
#pragma unroll
        for (int ni = 0; ni < 2; ++ni) {
            int row = m0 + mi * 16 + hh * 8;
            int col = n0 + ni * 16 + l;
#pragma unroll
            for (int r = 0; r < 8; ++r)
                C[(size_t)(row + r) * N + col] = acc[mi][ni][r];
        }
}

// -------------------------------------------------------------------------
// Kernel 4: sum split-K partials + batch-dim BatchNorm + ReLU, emit f16.
// One thread per column; reduction over Mrows (coalesced across threads).
// (FC bias cancels exactly inside (x - mean), so it is omitted upstream.)
// -------------------------------------------------------------------------
__global__ __launch_bounds__(256)
void bn_relu_kernel(const float* __restrict__ Xp,  // [KSPLIT][Mrows,Ncols]
                    const float* __restrict__ g, const float* __restrict__ be,
                    _Float16* __restrict__ Y, int Mrows, int Ncols)
{
    int j = blockIdx.x * blockDim.x + threadIdx.x;
    if (j >= Ncols) return;
    const size_t MN = (size_t)Mrows * Ncols;
    float s = 0.0f, ss = 0.0f;
    for (int i = 0; i < Mrows; ++i) {
        size_t o = (size_t)i * Ncols + j;
        float v = Xp[o] + Xp[o + MN] + Xp[o + 2 * MN] + Xp[o + 3 * MN];
        s += v; ss += v * v;
    }
    float mu  = s / (float)Mrows;
    float var = ss / (float)Mrows - mu * mu;
    float sc  = rsqrtf(var + 1e-5f) * g[j];
    float sh  = be[j];
    for (int i = 0; i < Mrows; ++i) {
        size_t o = (size_t)i * Ncols + j;
        float v = Xp[o] + Xp[o + MN] + Xp[o + 2 * MN] + Xp[o + 3 * MN];
        v = (v - mu) * sc + sh;
        v = fmaxf(v, 0.0f);
        Y[o] = (_Float16)v;
    }
}

// -------------------------------------------------------------------------
// Kernel 5: prediction head (93 dot products of length 1024) + postprocess.
// One block per box.
// -------------------------------------------------------------------------
__global__ __launch_bounds__(128)
void pred_post_kernel(const _Float16* __restrict__ act, const float* __restrict__ Wp,
                      const float* __restrict__ bp, const float* __restrict__ bbox,
                      const int* __restrict__ anchor, const float* __restrict__ zpad,
                      float* __restrict__ out)
{
    __shared__ float sp[OUTCH];
    const int m  = blockIdx.x;
    const int t  = threadIdx.x;
    const float zp = zpad[m];

    if (t < OUTCH) {
        const _Float16* ar = act + (size_t)m * FCDIM;
        const float*    wr = Wp  + (size_t)t * FCDIM;
        float s = 0.0f;
        for (int k = 0; k < FCDIM; k += 8) {
            v8h a = *(const v8h*)(ar + k);
#pragma unroll
            for (int u = 0; u < 8; ++u) s += (float)a[u] * wr[k + u];
        }
        sp[t] = (s + bp[t]) * zp;
    }
    __syncthreads();

    float* orow = out + (size_t)m * OUTCH;
    const int   level   = anchor[m] / 3;
    const float strides = (float)level * zp;
    const float st      = 4.0f * exp2f(strides);
    const float ay = bbox[m * 4 + 0], ax = bbox[m * 4 + 1];
    const float ah = bbox[m * 4 + 2], aw = bbox[m * 4 + 3];

    if (t < 4) {
        // bbox3d for class t: [cy, cx, z, nh, nw, ln]
        float d0 = sp[5 + t * 6 + 0], d1 = sp[5 + t * 6 + 1];
        float d2 = sp[5 + t * 6 + 2], d3 = sp[5 + t * 6 + 3];
        float d4 = sp[5 + t * 6 + 4], d5 = sp[5 + t * 6 + 5];
        orow[t * 6 + 0] = ay + d0 * ah;
        orow[t * 6 + 1] = ax + d1 * aw;
        orow[t * 6 + 2] = d4 * st;
        orow[t * 6 + 3] = ah * expf(d2);
        orow[t * 6 + 4] = aw * expf(d3);
        orow[t * 6 + 5] = expf(d5) * st;
        // yaw_cls softmax, row t (8 bins)
        float mx = -1e30f;
        for (int i = 0; i < 8; ++i) mx = fmaxf(mx, sp[29 + t * 8 + i]);
        float e[8], se = 0.0f;
        for (int i = 0; i < 8; ++i) { e[i] = expf(sp[29 + t * 8 + i] - mx); se += e[i]; }
        for (int i = 0; i < 8; ++i) orow[29 + t * 8 + i] = e[i] / se;
        // yaw_res: sigmoid * 0.6 - 0.3
        for (int i = 0; i < 8; ++i) {
            float v = sp[61 + t * 8 + i];
            orow[61 + t * 8 + i] = (1.0f / (1.0f + expf(-v))) * 0.6f - 0.3f;
        }
    } else if (t == 4) {
        // category softmax (5 logits) -> out[24..28]
        float mx = -1e30f;
        for (int i = 0; i < 5; ++i) mx = fmaxf(mx, sp[i]);
        float e[5], se = 0.0f;
        for (int i = 0; i < 5; ++i) { e[i] = expf(sp[i] - mx); se += e[i]; }
        for (int i = 0; i < 5; ++i) orow[24 + i] = e[i] / se;
    }
}

// -------------------------------------------------------------------------
// Launch
// -------------------------------------------------------------------------
extern "C" void kernel_launch(void* const* d_in, const int* in_sizes, int n_in,
                              void* d_out, int out_size, void* d_ws, size_t ws_size,
                              hipStream_t stream)
{
    (void)in_sizes; (void)n_in; (void)out_size; (void)ws_size;

    const float* f0     = (const float*)d_in[0];
    const float* f1     = (const float*)d_in[1];
    const float* f2     = (const float*)d_in[2];
    const float* f3     = (const float*)d_in[3];
    const float* bbox   = (const float*)d_in[4];
    const int*   anchor = (const int*)d_in[5];
    const float* zpad   = (const float*)d_in[6];
    const float* fc1_w  = (const float*)d_in[7];
    // d_in[8] fc1_b: cancels through BN (exact), omitted
    const float* bn1_g  = (const float*)d_in[9];
    const float* bn1_b  = (const float*)d_in[10];
    const float* fc2_w  = (const float*)d_in[11];
    // d_in[12] fc2_b: cancels through BN (exact), omitted
    const float* bn2_g  = (const float*)d_in[13];
    const float* bn2_b  = (const float*)d_in[14];
    const float* pred_w = (const float*)d_in[15];
    const float* pred_b = (const float*)d_in[16];

    char* ws = (char*)d_ws;
    _Float16* pooled = (_Float16*)(ws + 0);          // 512*12544*2  = 12,845,056
    _Float16* w1h    = (_Float16*)(ws + 12845056);   // 12544*1024*2 = 25,690,112
    float*    gP     = (float*)   (ws + 38535168);   // 4*512*1024*4 = 8,388,608 (reused FC1/FC2)
    _Float16* a1     = (_Float16*)(ws + 46923776);   // 512*1024*2   = 1,048,576
    _Float16* w2h    = (_Float16*)(ws + 47972352);   // 1024*1024*2  = 2,097,152
    _Float16* a2     = (_Float16*)(ws + 50069504);   // 512*1024*2   = 1,048,576
    float*    out    = (float*)d_out;

    const int M = BATCH * NBOX;   // 512

    roi_pool_kernel<<<M, 256, 0, stream>>>(f0, f1, f2, f3, bbox, anchor, pooled);
    cvt_f32_to_f16<<<2048, 256, 0, stream>>>(fc1_w, w1h, K1 * FCDIM);
    cvt_f32_to_f16<<<512, 256, 0, stream>>>(fc2_w, w2h, FCDIM * FCDIM);

    dim3 gg(FCDIM / 32, M / 32, KSPLIT);  // (32, 16, 4) -> 2048 waves
    gemm_f16_wmma<K1, K1 / KSPLIT, FCDIM><<<gg, 32, 0, stream>>>(pooled, w1h, gP, M);
    bn_relu_kernel<<<FCDIM / 256, 256, 0, stream>>>(gP, bn1_g, bn1_b, a1, M, FCDIM);
    gemm_f16_wmma<FCDIM, FCDIM / KSPLIT, FCDIM><<<gg, 32, 0, stream>>>(a1, w2h, gP, M);
    bn_relu_kernel<<<FCDIM / 256, 256, 0, stream>>>(gP, bn2_g, bn2_b, a2, M, FCDIM);

    pred_post_kernel<<<M, 128, 0, stream>>>(a2, pred_w, pred_b, bbox, anchor, zpad, out);
}
